// HCoN_58437325030126
// MI455X (gfx1250) — compile-verified, matrix operand order checked by reading, not run on hardware
//
#include <hip/hip_runtime.h>
#include <hip/hip_bf16.h>

// ---------------------------------------------------------------------------
// HCoN layer for MI455X (gfx1250, wave32, WMMA)
//   N=65536 nodes, E=16384 hyperedges, NNZ=786432, F=128
// ---------------------------------------------------------------------------

#define N_NODES  65536
#define N_EDGES  16384
#define NNZ_CNT  786432
#define FDIM     128

typedef __attribute__((ext_vector_type(16))) _Float16 v16h;
typedef __attribute__((ext_vector_type(8)))  _Float16 v8h;
typedef __attribute__((ext_vector_type(8)))  float    v8f;

// -------------------------------- f32 -> f16 --------------------------------
__global__ __launch_bounds__(256) void cvt_f32_f16(const float* __restrict__ in,
                                                   _Float16* __restrict__ out,
                                                   long n) {
    long i = (long)blockIdx.x * blockDim.x + threadIdx.x;
    if (i < n) out[i] = (_Float16)in[i];
}

// ------------------------------ scaled copy / zero ---------------------------
__global__ __launch_bounds__(256) void scale_copy_f32(const float* __restrict__ in,
                                                      float* __restrict__ out,
                                                      float s, long n) {
    long i = (long)blockIdx.x * blockDim.x + threadIdx.x;
    if (i < n) out[i] = s * in[i];
}

__global__ __launch_bounds__(256) void zero_f32(float* __restrict__ p, long n) {
    long i = (long)blockIdx.x * blockDim.x + threadIdx.x;
    if (i < n) p[i] = 0.0f;
}

// ---------------------------------------------------------------------------
// GEMM:  C[M,128] = A[M,128] (f16) @ W[128,128]^T (f16, row-major [out,in]) + bias
//
// One wave32 computes a 16x16 tile of C via v_wmma_f32_16x16x32_f16.
// Block = 256 threads = 8 waves -> the 8 column tiles of a 16-row panel.
//
// 16-bit A-fragment layout (ISA 7.12.2): lanes 0-15 hold row M=lane with
// K chunks [k0,k0+8) and [k0+16,k0+24); lanes 16-31 hold the same rows with
// chunks [k0+8,k0+16) and [k0+24,k0+32).  B (K x 16, column-major in memory
// == rows of W) mirrors this with lane = output column, so both fragments are
// two contiguous 16-byte loads per lane.
// ---------------------------------------------------------------------------
__global__ __launch_bounds__(256) void gemm_wmma_f16(const _Float16* __restrict__ A,
                                                     const _Float16* __restrict__ W,
                                                     const float* __restrict__ bias,
                                                     float* __restrict__ C,
                                                     int M) {
    const int lane    = threadIdx.x & 31;
    const int l15     = lane & 15;
    const int halfSel = lane >> 4;            // 0 or 1
    const int n0      = (threadIdx.x >> 5) * 16;   // wave id -> column tile
    const int m0      = blockIdx.x * 16;

    // accumulator seeded with bias (C layout: col = lane&15, row = i + 8*halfSel)
    float bv = bias[n0 + l15];
    v8f acc;
#pragma unroll
    for (int i = 0; i < 8; ++i) acc[i] = bv;

    const _Float16* arow = A + (long)(m0 + l15) * FDIM;
    const _Float16* wrow = W + (long)(n0 + l15) * FDIM;

#pragma unroll
    for (int k0 = 0; k0 < FDIM; k0 += 32) {
        const int ka = k0 + halfSel * 8;
        v8h a_lo = *(const v8h*)(arow + ka);
        v8h a_hi = *(const v8h*)(arow + ka + 16);
        v8h b_lo = *(const v8h*)(wrow + ka);
        v8h b_hi = *(const v8h*)(wrow + ka + 16);
        v16h afrag, bfrag;
#pragma unroll
        for (int i = 0; i < 8; ++i) {
            afrag[i]     = a_lo[i];
            afrag[i + 8] = a_hi[i];
            bfrag[i]     = b_lo[i];
            bfrag[i + 8] = b_hi[i];
        }
        acc = __builtin_amdgcn_wmma_f32_16x16x32_f16(
            /*neg_a=*/false, afrag, /*neg_b=*/false, bfrag,
            /*c_mod=*/(short)0, acc, /*reuse_a=*/false, /*reuse_b=*/false);
    }

    float* cbase = C + (long)(m0 + halfSel * 8) * FDIM + n0 + l15;
#pragma unroll
    for (int i = 0; i < 8; ++i) cbase[(long)i * FDIM] = acc[i];
}

// ---------------------------------------------------------------------------
// Segment-sum scatter:  dst[sidx[e], :] += scale * vals[e] * src[gidx[e], :]
// One wave per nonzero: float4 gather per lane + 4 global_atomic_add_f32.
// ---------------------------------------------------------------------------
__global__ __launch_bounds__(256) void spmm_scatter(const float* __restrict__ vals,
                                                    const int* __restrict__ gidx,
                                                    const int* __restrict__ sidx,
                                                    const float* __restrict__ src,
                                                    float* __restrict__ dst,
                                                    float scale, int nnz) {
    const int e    = (int)(((long)blockIdx.x * blockDim.x + threadIdx.x) >> 5);
    const int lane = threadIdx.x & 31;
    if (e >= nnz) return;

    const float v = scale * vals[e];
    const float4* srow = (const float4*)(src + (long)gidx[e] * FDIM);
    float* drow = dst + (long)sidx[e] * FDIM + lane * 4;

    float4 d = srow[lane];
    atomicAdd(drow + 0, v * d.x);
    atomicAdd(drow + 1, v * d.y);
    atomicAdd(drow + 2, v * d.z);
    atomicAdd(drow + 3, v * d.w);
}

// ---------------------------------------------------------------------------
extern "C" void kernel_launch(void* const* d_in, const int* in_sizes, int n_in,
                              void* d_out, int out_size, void* d_ws, size_t ws_size,
                              hipStream_t stream) {
    const float* x         = (const float*)d_in[0];   // [N,F]
    const float* y         = (const float*)d_in[1];   // [E,F]
    const int*   node_idx  = (const int*)d_in[2];     // [NNZ]
    const int*   hedge_idx = (const int*)d_in[3];     // [NNZ]
    const float* hd_vals   = (const float*)d_in[4];
    const float* hb_vals   = (const float*)d_in[5];
    const float* dhwd_vals = (const float*)d_in[6];
    const float* bhub_vals = (const float*)d_in[7];
    const float* Qv_w = (const float*)d_in[8];
    const float* Qv_b = (const float*)d_in[9];
    const float* Qe_w = (const float*)d_in[10];
    const float* Qe_b = (const float*)d_in[11];
    const float* Pe_w = (const float*)d_in[12];
    const float* Pe_b = (const float*)d_in[13];
    const float* Pv_w = (const float*)d_in[14];
    const float* Pv_b = (const float*)d_in[15];

    const long NF = (long)N_NODES * FDIM;   // 8,388,608
    const long EF = (long)N_EDGES * FDIM;   // 2,097,152
    const long WW = (long)FDIM * FDIM;      // 16,384

    // ---- workspace carve-out ----
    char* ws = (char*)d_ws;
    _Float16* xh   = (_Float16*)ws;               ws += NF * sizeof(_Float16);
    _Float16* yh   = (_Float16*)ws;               ws += EF * sizeof(_Float16);
    _Float16* qvwh = (_Float16*)ws;               ws += WW * sizeof(_Float16);
    _Float16* qewh = (_Float16*)ws;               ws += WW * sizeof(_Float16);
    _Float16* pewh = (_Float16*)ws;               ws += WW * sizeof(_Float16);
    _Float16* pvwh = (_Float16*)ws;               ws += WW * sizeof(_Float16);
    float* qvx = (float*)ws;                      ws += NF * sizeof(float);
    float* qey = (float*)ws;                      ws += EF * sizeof(float);
    float* pey = (float*)ws;                      ws += EF * sizeof(float);
    float* pvx = (float*)ws;                      ws += NF * sizeof(float);
    float* t_e = (float*)ws;                      ws += EF * sizeof(float);
    float* t_n = (float*)ws;                      ws += NF * sizeof(float);

    float* X = (float*)d_out;        // [N,F]
    float* Y = X + NF;               // [E,F]

    const int BLK = 256;
    auto cdiv = [](long a, long b) { return (int)((a + b - 1) / b); };

    // ---- 1) precision conversion ----
    cvt_f32_f16<<<cdiv(NF, BLK), BLK, 0, stream>>>(x, xh, NF);
    cvt_f32_f16<<<cdiv(EF, BLK), BLK, 0, stream>>>(y, yh, EF);
    cvt_f32_f16<<<cdiv(WW, BLK), BLK, 0, stream>>>(Qv_w, qvwh, WW);
    cvt_f32_f16<<<cdiv(WW, BLK), BLK, 0, stream>>>(Qe_w, qewh, WW);
    cvt_f32_f16<<<cdiv(WW, BLK), BLK, 0, stream>>>(Pe_w, pewh, WW);
    cvt_f32_f16<<<cdiv(WW, BLK), BLK, 0, stream>>>(Pv_w, pvwh, WW);

    // ---- 2) WMMA GEMMs: linear(x|y, W, b) ----
    gemm_wmma_f16<<<N_NODES / 16, BLK, 0, stream>>>(xh, qvwh, Qv_b, qvx, N_NODES);
    gemm_wmma_f16<<<N_EDGES / 16, BLK, 0, stream>>>(yh, qewh, Qe_b, qey, N_EDGES);
    gemm_wmma_f16<<<N_EDGES / 16, BLK, 0, stream>>>(yh, pewh, Pe_b, pey, N_EDGES);
    gemm_wmma_f16<<<N_NODES / 16, BLK, 0, stream>>>(xh, pvwh, Pv_b, pvx, N_NODES);

    const int scatterBlocks = cdiv((long)NNZ_CNT * 32, BLK);

    // ---- 3) t_e = 0.5*qey + segsum_edges(0.5 * hd * qvx[node_idx]) ----
    scale_copy_f32<<<cdiv(EF, BLK), BLK, 0, stream>>>(qey, t_e, 0.5f, EF);
    spmm_scatter<<<scatterBlocks, BLK, 0, stream>>>(hd_vals, node_idx, hedge_idx,
                                                    qvx, t_e, 0.5f, NNZ_CNT);

    // ---- 4) X = segsum_nodes(dhwd * t_e[hedge_idx]) ----
    zero_f32<<<cdiv(NF, BLK), BLK, 0, stream>>>(X, NF);
    spmm_scatter<<<scatterBlocks, BLK, 0, stream>>>(dhwd_vals, hedge_idx, node_idx,
                                                    t_e, X, 1.0f, NNZ_CNT);

    // ---- 5) t_n = 0.5*pvx + segsum_nodes(0.5 * hb * pey[hedge_idx]) ----
    scale_copy_f32<<<cdiv(NF, BLK), BLK, 0, stream>>>(pvx, t_n, 0.5f, NF);
    spmm_scatter<<<scatterBlocks, BLK, 0, stream>>>(hb_vals, hedge_idx, node_idx,
                                                    pey, t_n, 0.5f, NNZ_CNT);

    // ---- 6) Y = segsum_edges(bhub * t_n[node_idx]) ----
    zero_f32<<<cdiv(EF, BLK), BLK, 0, stream>>>(Y, EF);
    spmm_scatter<<<scatterBlocks, BLK, 0, stream>>>(bhub_vals, node_idx, hedge_idx,
                                                    t_n, Y, 1.0f, NNZ_CNT);
}
